// DualMPNN_15805479649769
// MI455X (gfx1250) — matrix operand order repaired and checked by reading, not compiled, not run on hardware
//
#include <hip/hip_runtime.h>
#include <hip/hip_bf16.h>

// ---------------------------------------------------------------------------
// DualMPNN forward for MI455X (gfx1250, wave32).
// Dense GEMMs: V_WMMA_F32_16X16X32_F16, f16 staged in LDS via b128 global
// loads + packed 8B LDS stores, f32 accumulation.  Segment softmax and
// aggregation via f32 global atomics.
// ---------------------------------------------------------------------------

typedef __attribute__((ext_vector_type(16))) _Float16 v16h;
typedef __attribute__((ext_vector_type(4)))  _Float16 v4h;
typedef __attribute__((ext_vector_type(8)))  float    v8f;

#define kN 30000
#define kE 120000
#define kD 128
#define kH 8

// ---------------- WMMA helpers (CDNA5 VGPR layouts, wave32) -----------------

__device__ __forceinline__ v8f wmma_f32_f16(v16h a, v16h b, v8f c) {
  // (neg_a, A, neg_b, B, c_mod, C, reuse_a, reuse_b)
  return __builtin_amdgcn_wmma_f32_16x16x32_f16(false, a, false, b,
                                                (short)0, c, false, false);
}

// A fragment: 16x32 f16, rows [row0,row0+16), K columns [k0,k0+32) of an LDS
// tile with leading dimension lda (in halfs).  Layout per ISA 7.12.2:
// lanes 0-15 (M=lane): halfs 0-7 -> K 0-7, halfs 8-15 -> K 16-23
// lanes 16-31 (M=lane-16): halfs 0-7 -> K 8-15, halfs 8-15 -> K 24-31
__device__ __forceinline__ v16h load_A_frag(const _Float16* As, int lda,
                                            int row0, int k0, int lane) {
  v16h a;
  const int m    = lane & 15;
  const int koff = (lane & 16) ? 8 : 0;
  const _Float16* p = As + (row0 + m) * lda + k0 + koff;
#pragma unroll
  for (int j = 0; j < 8; ++j) a[j] = p[j];
#pragma unroll
  for (int j = 0; j < 8; ++j) a[8 + j] = p[16 + j];
  return a;
}

// B fragment: 32x16 f16, K rows [k0,k0+32), N columns [n0,n0+16).
// lanes 0-15 (N=lane): halfs 0-7 -> K 0-7, halfs 8-15 -> K 16-23
// lanes 16-31: halfs 0-7 -> K 8-15, halfs 8-15 -> K 24-31
__device__ __forceinline__ v16h load_B_frag(const _Float16* Bs, int ldb,
                                            int k0, int n0, int lane) {
  v16h b;
  const int n  = n0 + (lane & 15);
  const int kk = k0 + ((lane & 16) ? 8 : 0);
#pragma unroll
  for (int j = 0; j < 8; ++j) b[j] = Bs[(kk + j) * ldb + n];
#pragma unroll
  for (int j = 0; j < 8; ++j) b[8 + j] = Bs[(kk + 16 + j) * ldb + n];
  return b;
}
// C/D 16x16 f32: c[r] = D[r + (lane<16?0:8)][lane&15]

__device__ __forceinline__ v4h cvt4(float4 v) {
  v4h h;
  h[0] = (_Float16)v.x; h[1] = (_Float16)v.y;
  h[2] = (_Float16)v.z; h[3] = (_Float16)v.w;
  return h;
}

// ---------------- misc device helpers ----------------

__device__ __forceinline__ void atomicMaxF(float* addr, float val) {
  // ordered-int trick; init value must be a large negative float
  if (val >= 0.0f) atomicMax((int*)addr, __float_as_int(val));
  else             atomicMin((unsigned int*)addr, __float_as_uint(val));
}

// ---------------- kernels ----------------

// out[row,:] = sum_k tab[k, idx[row,k], :]
__global__ __launch_bounds__(256) void k_encode(const int* __restrict__ idx,
                                                const float* __restrict__ tab,
                                                float* __restrict__ out,
                                                int M, int K, int V) {
  int i = blockIdx.x * 256 + threadIdx.x;
  if (i >= M * kD) return;
  int row = i >> 7, c = i & 127;
  float s = 0.f;
  for (int k = 0; k < K; ++k) {
    int id = idx[row * K + k];
    s += tab[((size_t)k * V + id) * kD + c];
  }
  out[i] = s;
}

__global__ __launch_bounds__(256) void k_fill(float* __restrict__ p, float v, int n) {
  int i = blockIdx.x * 256 + threadIdx.x;
  if (i < n) p[i] = v;
}

// Y[M,128] = X[M,128] @ W[128,128]   (WMMA, 64 rows per block)
__global__ __launch_bounds__(256) void k_gemm_node(const float* __restrict__ X,
                                                   const float* __restrict__ W,
                                                   float* __restrict__ Y, int M) {
  constexpr int WLD = kD + 8;
  __shared__ _Float16 Xs[64 * WLD];
  __shared__ _Float16 Wt[32 * WLD];
  const int tid = threadIdx.x;
  const int wave = tid >> 5, lane = tid & 31;
  const int n0 = wave * 16;
  const int row0 = blockIdx.x * 64;
  const bool full = (row0 + 64 <= M);

  // stage X tile: float4 global loads, packed 4xf16 LDS stores
  if (full) {
#pragma unroll
    for (int it = 0; it < 8; ++it) {
      int idx = tid + it * 256;             // 2048 float4 total
      int r = idx >> 5, c4 = idx & 31;
      float4 v = reinterpret_cast<const float4*>(X + (size_t)(row0 + r) * kD)[c4];
      *reinterpret_cast<v4h*>(&Xs[r * WLD + c4 * 4]) = cvt4(v);
    }
  } else {
#pragma unroll
    for (int it = 0; it < 8; ++it) {
      int idx = tid + it * 256;
      int r = idx >> 5, c4 = idx & 31;
      int gr = row0 + r; if (gr >= M) gr = M - 1;     // clamp: EXEC stays full
      float4 v = reinterpret_cast<const float4*>(X + (size_t)gr * kD)[c4];
      *reinterpret_cast<v4h*>(&Xs[r * WLD + c4 * 4]) = cvt4(v);
    }
  }

  v8f acc[4] = {{}, {}, {}, {}};
  for (int ks = 0; ks < 4; ++ks) {
    __syncthreads();
#pragma unroll
    for (int it = 0; it < 4; ++it) {
      int idx = tid + it * 256;             // 1024 float4 total
      int r = idx >> 5, c4 = idx & 31;
      float4 v = reinterpret_cast<const float4*>(W + (size_t)(ks * 32 + r) * kD)[c4];
      *reinterpret_cast<v4h*>(&Wt[r * WLD + c4 * 4]) = cvt4(v);
    }
    __syncthreads();
    v16h b = load_B_frag(Wt, WLD, 0, n0, lane);
#pragma unroll
    for (int rt = 0; rt < 4; ++rt) {
      v16h a = load_A_frag(Xs, WLD, rt * 16, ks * 32, lane);
      acc[rt] = wmma_f32_f16(a, b, acc[rt]);
    }
  }
  const int n  = n0 + (lane & 15);
  const int mb = (lane & 16) ? 8 : 0;
  if (full) {
#pragma unroll
    for (int rt = 0; rt < 4; ++rt)
#pragma unroll
      for (int r = 0; r < 8; ++r)
        Y[(size_t)(row0 + rt * 16 + mb + r) * kD + n] = acc[rt][r];
  } else {
#pragma unroll
    for (int rt = 0; rt < 4; ++rt)
#pragma unroll
      for (int r = 0; r < 8; ++r) {
        int gr = row0 + rt * 16 + mb + r;
        if (gr < M) Y[(size_t)gr * kD + n] = acc[rt][r];
      }
  }
}

// Y[row,:] = ef[row,:] @ W, where ef[row] = E_in[row] for row<Me, else self_e.
__global__ __launch_bounds__(256) void k_gemm_ef(const float* __restrict__ Ein,
                                                 const float* __restrict__ Se,
                                                 const float* __restrict__ W,
                                                 float* __restrict__ Y,
                                                 int Mtot, int Me) {
  constexpr int WLD = kD + 8;
  __shared__ _Float16 Xs[64 * WLD];
  __shared__ _Float16 Wt[32 * WLD];
  const int tid = threadIdx.x;
  const int wave = tid >> 5, lane = tid & 31;
  const int n0 = wave * 16;
  const int row0 = blockIdx.x * 64;
  const bool full = (row0 + 64 <= Mtot);

#pragma unroll
  for (int it = 0; it < 8; ++it) {
    int idx = tid + it * 256;
    int r = idx >> 5, c4 = idx & 31;
    int gr = row0 + r; if (gr >= Mtot) gr = Mtot - 1;
    const float* base = (gr < Me) ? (Ein + (size_t)gr * kD) : Se;
    float4 v = reinterpret_cast<const float4*>(base)[c4];
    *reinterpret_cast<v4h*>(&Xs[r * WLD + c4 * 4]) = cvt4(v);
  }

  v8f acc[4] = {{}, {}, {}, {}};
  for (int ks = 0; ks < 4; ++ks) {
    __syncthreads();
#pragma unroll
    for (int it = 0; it < 4; ++it) {
      int idx = tid + it * 256;
      int r = idx >> 5, c4 = idx & 31;
      float4 v = reinterpret_cast<const float4*>(W + (size_t)(ks * 32 + r) * kD)[c4];
      *reinterpret_cast<v4h*>(&Wt[r * WLD + c4 * 4]) = cvt4(v);
    }
    __syncthreads();
    v16h b = load_B_frag(Wt, WLD, 0, n0, lane);
#pragma unroll
    for (int rt = 0; rt < 4; ++rt) {
      v16h a = load_A_frag(Xs, WLD, rt * 16, ks * 32, lane);
      acc[rt] = wmma_f32_f16(a, b, acc[rt]);
    }
  }
  const int n  = n0 + (lane & 15);
  const int mb = (lane & 16) ? 8 : 0;
  if (full) {
#pragma unroll
    for (int rt = 0; rt < 4; ++rt)
#pragma unroll
      for (int r = 0; r < 8; ++r)
        Y[(size_t)(row0 + rt * 16 + mb + r) * kD + n] = acc[rt][r];
  } else {
#pragma unroll
    for (int rt = 0; rt < 4; ++rt)
#pragma unroll
      for (int r = 0; r < 8; ++r) {
        int gr = row0 + rt * 16 + mb + r;
        if (gr < Mtot) Y[(size_t)gr * kD + n] = acc[rt][r];
      }
  }
}

// out[row,h] = sum_c X[row, h*16+c] * a[h*16+c]
__global__ __launch_bounds__(256) void k_rowdot(const float* __restrict__ X,
                                                const float* __restrict__ a,
                                                float* __restrict__ out, int M) {
  int i = blockIdx.x * 256 + threadIdx.x;
  if (i >= M * kH) return;
  int row = i >> 3, hh = i & 7;
  const float* xp = X + (size_t)row * kD + hh * 16;
  const float* ap = a + hh * 16;
  float s = 0.f;
#pragma unroll
  for (int c = 0; c < 16; ++c) s += xp[c] * ap[c];
  out[i] = s;
}

__global__ __launch_bounds__(256) void k_alpha_max(const int* __restrict__ esrc,
                                                   const int* __restrict__ edst,
                                                   const float* __restrict__ sv,
                                                   const float* __restrict__ dv,
                                                   float* __restrict__ aev,
                                                   float* __restrict__ amax,
                                                   int En, int Rows) {
  int i = blockIdx.x * 256 + threadIdx.x;
  if (i >= Rows * kH) return;
  int row = i >> 3, hh = i & 7;
  int src = (row < En) ? esrc[row] : row - En;   // self loops appended
  int dst = (row < En) ? edst[row] : row - En;
  float al = sv[src * kH + hh] + dv[dst * kH + hh] + aev[i];
  al = (al > 0.f) ? al : 0.2f * al;              // leaky_relu, NEG=0.2
  aev[i] = al;
  atomicMaxF(&amax[dst * kH + hh], al);
}

__global__ __launch_bounds__(256) void k_exp_den(const int* __restrict__ edst,
                                                 const float* __restrict__ amax,
                                                 float* __restrict__ aev,
                                                 float* __restrict__ den,
                                                 int En, int Rows) {
  int i = blockIdx.x * 256 + threadIdx.x;
  if (i >= Rows * kH) return;
  int row = i >> 3, hh = i & 7;
  int dst = (row < En) ? edst[row] : row - En;
  float ex = expf(aev[i] - amax[dst * kH + hh]);
  aev[i] = ex;
  atomicAdd(&den[dst * kH + hh], ex);
}

__global__ __launch_bounds__(256) void k_aggregate(const int* __restrict__ esrc,
                                                   const int* __restrict__ edst,
                                                   const float* __restrict__ aev,
                                                   const float* __restrict__ den,
                                                   const float* __restrict__ hbuf,
                                                   float* __restrict__ acc,
                                                   int En, int Rows) {
  int i = blockIdx.x * 256 + threadIdx.x;
  if (i >= Rows * kH) return;
  int row = i >> 3, hh = i & 7;
  int src = (row < En) ? esrc[row] : row - En;
  int dst = (row < En) ? edst[row] : row - En;
  float w = aev[i] / den[dst * kH + hh];
  const float* hp = hbuf + (size_t)src * kD + hh * 16;
  float* ap = acc + (size_t)dst * kD + hh * 16;
#pragma unroll
  for (int c = 0; c < 16; ++c) atomicAdd(ap + c, hp[c] * w);
}

// x[row] = relu(LN(acc[row]+bias)*w+b) + x[row]   (one block per row)
__global__ __launch_bounds__(128) void k_ln_node(const float* __restrict__ acc,
                                                 const float* __restrict__ bias,
                                                 const float* __restrict__ w,
                                                 const float* __restrict__ b,
                                                 float* __restrict__ x, int M) {
  __shared__ float red[128];
  int row = blockIdx.x;
  if (row >= M) return;
  int t = threadIdx.x;
  float v = acc[(size_t)row * kD + t] + bias[t];
  red[t] = v; __syncthreads();
  for (int off = 64; off > 0; off >>= 1) {
    if (t < off) red[t] += red[t + off];
    __syncthreads();
  }
  float mu = red[0] * (1.f / 128.f); __syncthreads();
  float dd = v - mu;
  red[t] = dd * dd; __syncthreads();
  for (int off = 64; off > 0; off >>= 1) {
    if (t < off) red[t] += red[t + off];
    __syncthreads();
  }
  float var = red[0] * (1.f / 128.f);
  float y = dd * rsqrtf(var + 1e-5f) * w[t] + b[t];
  y = (y > 0.f) ? y : 0.f;
  x[(size_t)row * kD + t] = y + x[(size_t)row * kD + t];
}

// Fused edge MLP: e = relu(LN(relu([x_s|x_d|e]@W1+b1)@W2+b2)) + e
// 16 edges per block; two WMMA GEMMs (K=384, K=128) fully in LDS.
__global__ __launch_bounds__(256) void k_edge_update(
    const float* __restrict__ x, const int* __restrict__ esrc,
    const int* __restrict__ edst, float* __restrict__ e,
    const float* __restrict__ w1, const float* __restrict__ b1,
    const float* __restrict__ w2, const float* __restrict__ b2,
    const float* __restrict__ lnw, const float* __restrict__ lnb, int En) {
  constexpr int ZLD = 3 * kD + 8;
  constexpr int WLD = kD + 8;
  __shared__ _Float16 Zs[16 * ZLD];
  __shared__ _Float16 Wt[32 * WLD];
  __shared__ _Float16 H1s[16 * WLD];
  __shared__ float Yt[16 * kD];
  __shared__ float red[16 * 16];

  const int tid = threadIdx.x;
  const int wave = tid >> 5, lane = tid & 31;
  const int n0 = wave * 16;
  const int e0 = blockIdx.x * 16;

  // gather [x_src | x_dst | e] tile (float4 loads; each float4 stays inside
  // one 128-wide segment)
#pragma unroll
  for (int it = 0; it < 6; ++it) {
    int idx = tid + it * 256;               // 16*96 = 1536 float4 total
    int r = idx / 96, c4 = idx - r * 96;
    int ge = e0 + r; if (ge >= En) ge = En - 1;
    int seg = c4 >> 5, off4 = c4 & 31;
    const float* base;
    if (seg == 0)      base = x + (size_t)esrc[ge] * kD;
    else if (seg == 1) base = x + (size_t)edst[ge] * kD;
    else               base = e + (size_t)ge * kD;
    float4 v = reinterpret_cast<const float4*>(base)[off4];
    *reinterpret_cast<v4h*>(&Zs[r * ZLD + c4 * 4]) = cvt4(v);
  }

  // GEMM1: 16x384 @ 384x128
  v8f c1 = {};
  for (int ks = 0; ks < 12; ++ks) {
    __syncthreads();
#pragma unroll
    for (int it = 0; it < 4; ++it) {
      int idx = tid + it * 256;
      int r = idx >> 5, c4 = idx & 31;
      float4 v = reinterpret_cast<const float4*>(w1 + (size_t)(ks * 32 + r) * kD)[c4];
      *reinterpret_cast<v4h*>(&Wt[r * WLD + c4 * 4]) = cvt4(v);
    }
    __syncthreads();
    v16h a = load_A_frag(Zs, ZLD, 0, ks * 32, lane);
    v16h b = load_B_frag(Wt, WLD, 0, n0, lane);
    c1 = wmma_f32_f16(a, b, c1);
  }
  __syncthreads();
  {
    const int n = n0 + (lane & 15);
    const int mb = (lane & 16) ? 8 : 0;
    const float bb = b1[n];
#pragma unroll
    for (int r = 0; r < 8; ++r) {
      float v = c1[r] + bb;
      v = (v > 0.f) ? v : 0.f;                 // relu
      H1s[(mb + r) * WLD + n] = (_Float16)v;
    }
  }
  __syncthreads();

  // GEMM2: 16x128 @ 128x128
  v8f c2 = {};
  for (int ks = 0; ks < 4; ++ks) {
#pragma unroll
    for (int it = 0; it < 4; ++it) {
      int idx = tid + it * 256;
      int r = idx >> 5, c4 = idx & 31;
      float4 v = reinterpret_cast<const float4*>(w2 + (size_t)(ks * 32 + r) * kD)[c4];
      *reinterpret_cast<v4h*>(&Wt[r * WLD + c4 * 4]) = cvt4(v);
    }
    __syncthreads();
    v16h a = load_A_frag(H1s, WLD, 0, ks * 32, lane);
    v16h b = load_B_frag(Wt, WLD, 0, n0, lane);
    c2 = wmma_f32_f16(a, b, c2);
    __syncthreads();
  }
  {
    const int n = n0 + (lane & 15);
    const int mb = (lane & 16) ? 8 : 0;
    const float bb = b2[n];
#pragma unroll
    for (int r = 0; r < 8; ++r) Yt[(mb + r) * kD + n] = c2[r] + bb;
  }
  __syncthreads();

  // LayerNorm + relu + residual: 16 threads per row
  const int row = tid >> 4, sub = tid & 15;
  float ps = 0.f;
#pragma unroll
  for (int j = 0; j < 8; ++j) ps += Yt[row * kD + j * 16 + sub];
  red[row * 16 + sub] = ps; __syncthreads();
  for (int off = 8; off > 0; off >>= 1) {
    if (sub < off) red[row * 16 + sub] += red[row * 16 + sub + off];
    __syncthreads();
  }
  const float mu = red[row * 16] * (1.f / 128.f); __syncthreads();
  float vs = 0.f;
#pragma unroll
  for (int j = 0; j < 8; ++j) {
    float dd = Yt[row * kD + j * 16 + sub] - mu;
    vs += dd * dd;
  }
  red[row * 16 + sub] = vs; __syncthreads();
  for (int off = 8; off > 0; off >>= 1) {
    if (sub < off) red[row * 16 + sub] += red[row * 16 + sub + off];
    __syncthreads();
  }
  const float rstd = rsqrtf(red[row * 16] * (1.f / 128.f) + 1e-5f);
  const int ge = e0 + row;
  if (ge < En) {
#pragma unroll
    for (int j = 0; j < 8; ++j) {
      int cc = j * 16 + sub;
      float y = (Yt[row * kD + cc] - mu) * rstd * lnw[cc] + lnb[cc];
      y = (y > 0.f) ? y : 0.f;
      e[(size_t)ge * kD + cc] = y + e[(size_t)ge * kD + cc];
    }
  }
}

// ---------------------------------------------------------------------------

extern "C" void kernel_launch(void* const* d_in, const int* in_sizes, int n_in,
                              void* d_out, int out_size, void* d_ws, size_t ws_size,
                              hipStream_t stream) {
  (void)in_sizes; (void)n_in; (void)out_size; (void)ws_size;

  const int*   reac_x_idx  = (const int*)d_in[0];
  const int*   reac_ei     = (const int*)d_in[1];
  const int*   reac_ea_idx = (const int*)d_in[2];
  const int*   prod_x_idx  = (const int*)d_in[3];
  const int*   prod_ei     = (const int*)d_in[4];
  const int*   prod_ea_idx = (const int*)d_in[5];
  const float* atom_emb  = (const float*)d_in[6];
  const float* bond_emb  = (const float*)d_in[7];
  const float* Wn        = (const float*)d_in[8];
  const float* We        = (const float*)d_in[9];
  const float* att_src   = (const float*)d_in[10];
  const float* att_dst   = (const float*)d_in[11];
  const float* att_edge  = (const float*)d_in[12];
  const float* conv_bias = (const float*)d_in[13];
  const float* self_edge = (const float*)d_in[14];
  const float* bn_w      = (const float*)d_in[15];
  const float* bn_b      = (const float*)d_in[16];
  const float* eu_w1     = (const float*)d_in[17];
  const float* eu_b1     = (const float*)d_in[18];
  const float* eu_w2     = (const float*)d_in[19];
  const float* eu_b2     = (const float*)d_in[20];
  const float* ln_w      = (const float*)d_in[21];
  const float* ln_b      = (const float*)d_in[22];

  // outputs double as the in-place feature buffers
  float* out    = (float*)d_out;
  float* reac_x = out;
  float* prod_x = out + (size_t)kN * kD;
  float* reac_e = out + (size_t)2 * kN * kD;
  float* prod_e = reac_e + (size_t)kE * kD;

  // workspace carve-out (floats): ~115 MB total
  float* ws    = (float*)d_ws;
  float* hbuf  = ws;                                  // [N,128]
  float* hebuf = hbuf + (size_t)kN * kD;              // [E+N,128]
  float* accb  = hebuf + (size_t)(kE + kN) * kD;      // [N,128]
  float* sv    = accb + (size_t)kN * kD;              // [N,8]
  float* dv    = sv + (size_t)kN * kH;                // [N,8]
  float* aev   = dv + (size_t)kN * kH;                // [E+N,8]
  float* amax  = aev + (size_t)(kE + kN) * kH;        // [N,8]
  float* den   = amax + (size_t)kN * kH;              // [N,8]

  const int R = kE + kN;

  // ---- encoders ----
  k_encode<<<(kN * kD + 255) / 256, 256, 0, stream>>>(reac_x_idx, atom_emb, reac_x, kN, 9, 64);
  k_encode<<<(kN * kD + 255) / 256, 256, 0, stream>>>(prod_x_idx, atom_emb, prod_x, kN, 9, 64);
  k_encode<<<(kE * kD + 255) / 256, 256, 0, stream>>>(reac_ea_idx, bond_emb, reac_e, kE, 3, 8);
  k_encode<<<(kE * kD + 255) / 256, 256, 0, stream>>>(prod_ea_idx, bond_emb, prod_e, kE, 3, 8);

  for (int layer = 0; layer < 2; ++layer) {
    // ---- GAT convs (reac then prod) ----
    for (int g = 0; g < 2; ++g) {
      const int li = 2 * layer + g;
      float* x = g ? prod_x : reac_x;
      float* e = g ? prod_e : reac_e;
      const int* esrc = g ? prod_ei : reac_ei;
      const int* edst = esrc + kE;

      k_gemm_node<<<(kN + 63) / 64, 256, 0, stream>>>(x, Wn + (size_t)li * kD * kD, hbuf, kN);
      k_gemm_ef<<<(R + 63) / 64, 256, 0, stream>>>(e, self_edge + li * kD,
                                                   We + (size_t)li * kD * kD, hebuf, R, kE);
      k_rowdot<<<(kN * kH + 255) / 256, 256, 0, stream>>>(hbuf, att_src + li * kD, sv, kN);
      k_rowdot<<<(kN * kH + 255) / 256, 256, 0, stream>>>(hbuf, att_dst + li * kD, dv, kN);
      k_rowdot<<<(R * kH + 255) / 256, 256, 0, stream>>>(hebuf, att_edge + li * kD, aev, R);

      k_fill<<<(kN * kH + 255) / 256, 256, 0, stream>>>(amax, -3.0e38f, kN * kH);
      k_fill<<<(kN * kH + 255) / 256, 256, 0, stream>>>(den, 0.f, kN * kH);
      k_fill<<<(kN * kD + 255) / 256, 256, 0, stream>>>(accb, 0.f, kN * kD);

      k_alpha_max<<<(R * kH + 255) / 256, 256, 0, stream>>>(esrc, edst, sv, dv, aev, amax, kE, R);
      k_exp_den<<<(R * kH + 255) / 256, 256, 0, stream>>>(edst, amax, aev, den, kE, R);
      k_aggregate<<<(R * kH + 255) / 256, 256, 0, stream>>>(esrc, edst, aev, den, hbuf, accb, kE, R);

      k_ln_node<<<kN, 128, 0, stream>>>(accb, conv_bias + li * kD,
                                        bn_w + li * kD, bn_b + li * kD, x, kN);
    }
    // ---- edge updates (use updated x, old e) ----
    for (int g = 0; g < 2; ++g) {
      const int li = 2 * layer + g;
      float* x = g ? prod_x : reac_x;
      float* e = g ? prod_e : reac_e;
      const int* esrc = g ? prod_ei : reac_ei;
      const int* edst = esrc + kE;
      k_edge_update<<<(kE + 15) / 16, 256, 0, stream>>>(
          x, esrc, edst, e,
          eu_w1 + (size_t)li * 3 * kD * kD, eu_b1 + li * kD,
          eu_w2 + (size_t)li * kD * kD, eu_b2 + li * kD,
          ln_w + li * kD, ln_b + li * kD, kE);
    }
  }
}
